// CLCRNModel_53085795778856
// MI455X (gfx1250) — matrix operand order repaired and checked by reading, not compiled, not run on hardware
//
#include <hip/hip_runtime.h>

// ---------------- static config (matches reference) ----------------
#define N_NODES 2048
#define NBHD    25
#define NEDGE   (N_NODES * NBHD)      // 51200
#define BATCH   8
#define SEQ     12
#define HOR     12
#define EMB     16
#define UNITS   64
#define M_ROWS  (BATCH * N_NODES)     // 16384
#define DENC    33                    // 2*EMB + 1
#define CENC    (DENC + UNITS)        // 97
#define KENC    (3 * CENC)            // 291
#define KPAD_E  320                   // pad to multiple of 32
#define CDEC    (1 + UNITS)           // 65
#define KDEC    (3 * CDEC)            // 195
#define KPAD_D  224

typedef _Float16 half_t;
typedef _Float16 v16h __attribute__((ext_vector_type(16)));
typedef _Float16 v8h  __attribute__((ext_vector_type(8)));
typedef float    v8f  __attribute__((ext_vector_type(8)));

// ---------------- utility ----------------
__global__ void k_zero_f32(float* p, int n) {
    int i = blockIdx.x * blockDim.x + threadIdx.x;
    if (i < n) p[i] = 0.f;
}

// ---------------- static edge weights: conditional-kernel MLP ----------------
__global__ void k_edge_mlp(const float* __restrict__ loc, const float* __restrict__ geo,
                           const float* __restrict__ ang,
                           const float* __restrict__ Wk0, const float* __restrict__ bk0,
                           const float* __restrict__ Wk1, const float* __restrict__ bk1,
                           const float* __restrict__ Wk2, const float* __restrict__ bk2,
                           const int* __restrict__ sidx, float* __restrict__ wraw) {
    int e = blockIdx.x * blockDim.x + threadIdx.x;
    if (e >= NEDGE) return;
    int r = sidx[e], c = sidx[NEDGE + e];
    float d0 = loc[2 * c] - loc[2 * r];
    float d1 = loc[2 * c + 1] - loc[2 * r + 1];
    float h0[32];
#pragma unroll
    for (int j = 0; j < 32; ++j)
        h0[j] = tanhf(d0 * Wk0[j] + d1 * Wk0[32 + j] + bk0[j]);
    float h1[16];
#pragma unroll
    for (int j = 0; j < 16; ++j) {
        float s = bk1[j];
#pragma unroll
        for (int i = 0; i < 32; ++i) s += h0[i] * Wk1[i * 16 + j];
        h1[j] = tanhf(s);
    }
    float s = bk2[0];
#pragma unroll
    for (int i = 0; i < 16; ++i) s += h1[i] * Wk2[i];
    float ker = (s > 20.f) ? s : log1pf(expf(s));   // softplus
    float g = geo[e];
    wraw[e] = ker * ang[e] * expf(-g * g);
}

// row = repeat(arange(N), 25) -> each node owns 25 contiguous edges: no atomics
__global__ void k_edge_norm(const float* __restrict__ wraw, float* __restrict__ w) {
    int n = blockIdx.x * blockDim.x + threadIdx.x;
    if (n >= N_NODES) return;
    float s = 0.f;
#pragma unroll
    for (int j = 0; j < NBHD; ++j) s += wraw[n * NBHD + j];
    float inv = 1.f / (s + 1e-8f);
#pragma unroll
    for (int j = 0; j < NBHD; ++j) w[n * NBHD + j] = wraw[n * NBHD + j] * inv;
}

// ---------------- weights: f32 (K,N) -> f16 transposed padded (N, KPAD) ----------------
__global__ void k_wtrans(const float* __restrict__ W, half_t* __restrict__ Wt,
                         int K, int NOUT, int KPAD) {
    int idx = blockIdx.x * blockDim.x + threadIdx.x;
    if (idx >= NOUT * KPAD) return;
    int n = idx / KPAD, k = idx % KPAD;
    Wt[idx] = (k < K) ? (half_t)W[k * NOUT + n] : (half_t)0.f;
}

// ---------------- pack: z = [feat_emb | node_emb | x | (r*)h], zero K-pad ----------------
__global__ void k_pack_enc(const float* __restrict__ inputs, const float* __restrict__ node_emb,
                           const float* __restrict__ W_fe, const float* __restrict__ b_fe,
                           const float* __restrict__ h, const float* __restrict__ ru,
                           int useR, int step, half_t* __restrict__ y) {
    int kk = threadIdx.x;
    int n = blockIdx.y, b = blockIdx.z;
    int m = b * N_NODES + n;
    if (kk >= CENC + (KPAD_E - KENC)) return;   // 126 active lanes
    float x = inputs[((size_t)step * BATCH + b) * N_NODES + n];
    float v; int k;
    if (kk < EMB)            { v = x * W_fe[kk] + b_fe[kk];            k = kk; }
    else if (kk < 2 * EMB)   { v = node_emb[n * EMB + (kk - EMB)];     k = kk; }
    else if (kk == 2 * EMB)  { v = x;                                  k = kk; }
    else if (kk < CENC)      { int j = kk - (2 * EMB + 1);
                               v = h[(size_t)m * UNITS + j];
                               if (useR) v *= ru[(size_t)m * 128 + j]; k = kk; }
    else                     { v = 0.f; k = KENC + (kk - CENC); }      // zero K-pad
    y[(size_t)m * KPAD_E + k] = (half_t)v;
}

__global__ void k_pack_dec(const float* __restrict__ dec_in, const float* __restrict__ h,
                           const float* __restrict__ ru, int useR, half_t* __restrict__ y) {
    int kk = threadIdx.x;
    int n = blockIdx.y, b = blockIdx.z;
    int m = b * N_NODES + n;
    if (kk >= CDEC + (KPAD_D - KDEC)) return;   // 94 active lanes
    float v; int k;
    if (kk == 0)        { v = dec_in[m]; k = 0; }
    else if (kk < CDEC) { int j = kk - 1;
                          v = h[(size_t)m * UNITS + j];
                          if (useR) v *= ru[(size_t)m * 128 + j]; k = kk; }
    else                { v = 0.f; k = KDEC + (kk - CDEC); }
    y[(size_t)m * KPAD_D + k] = (half_t)v;
}

// ---------------- sparse local conv hop (per-node gather-reduce, LDS-staged) -------------
__global__ void k_hop(half_t* __restrict__ y, const float* __restrict__ w,
                      const int* __restrict__ col, int C, int kin, int kout, int KPAD) {
    __shared__ int   scol[NBHD];
    __shared__ float sw[NBHD];
    int n = blockIdx.y, b = blockIdx.z;
    if (threadIdx.x < NBHD) {
        scol[threadIdx.x] = col[n * NBHD + threadIdx.x];
        sw[threadIdx.x]   = w[n * NBHD + threadIdx.x];
    }
    __syncthreads();
    int c = blockIdx.x * blockDim.x + threadIdx.x;
    if (c >= C) return;
    float acc = 0.f;
#pragma unroll
    for (int j = 0; j < NBHD; ++j)
        acc += sw[j] * (float)y[((size_t)(b * N_NODES + scol[j])) * KPAD + kin + c];
    y[((size_t)(b * N_NODES + n)) * KPAD + kout + c] = (half_t)acc;
}

// ---------------- WMMA GEMM + fused activation --------------------------------
// 8 waves/block, one 16-row M tile per wave. B panel staged through LDS in
// 64-column chunks (ds_load breaks the loadcnt serialization; B global traffic
// amortized over the whole block). A fragments held in registers per K step.
// MODE 0: ru = sigmoid(A@B + bias)            (NOUT=128)
// MODE 1: h  = u*h + (1-u)*tanh(A@B + bias)   (NOUT=64, u = ru[:,64:128])
template <int KPAD, int NOUT, int MODE>
__global__ void __launch_bounds__(256)
k_wmma_gemm(const half_t* __restrict__ A, const half_t* __restrict__ Bt,
            const float* __restrict__ bias, float* __restrict__ ru,
            float* __restrict__ h) {
    constexpr int LROW = KPAD + 8;            // LDS row stride (bank-scatter pad)
    constexpr int NCH  = NOUT / 64;           // 64-column chunks (4 N tiles each)
    __shared__ __align__(16) half_t Bs[64 * LROW];

    const int lane = threadIdx.x & 31;
    const int wave = threadIdx.x >> 5;
    const int m0   = (blockIdx.x * 8 + wave) * 16;
    const int hi   = lane >> 4;      // 0/1 lane half
    const int l15  = lane & 15;
    // A fragment: row m0+l15; 16 contiguous K per lane half: hi*16 + {0..15}
    const half_t* pA = A + (size_t)(m0 + l15) * KPAD + hi * 16;
    const int mr = m0 + hi * 8;      // C layout: VGPR j -> row mr+j, col n

    for (int ch = 0; ch < NCH; ++ch) {
        __syncthreads();             // protect Bs from previous chunk's readers
        // cooperative stage: 64 rows of Bt (cols ch*64 .. ch*64+63) -> LDS
        for (int i = threadIdx.x; i < 64 * KPAD / 8; i += 256) {
            int n = (i * 8) / KPAD, k = (i * 8) % KPAD;
            *(v8h*)&Bs[n * LROW + k] =
                *(const v8h*)(Bt + (size_t)(ch * 64 + n) * KPAD + k);
        }
        __syncthreads();

        v8f acc[4];
#pragma unroll
        for (int nt = 0; nt < 4; ++nt) acc[nt] = v8f{};

#pragma unroll
        for (int kt = 0; kt < KPAD / 32; ++kt) {
            v8h alo = *(const v8h*)(pA + kt * 32);
            v8h ahi = *(const v8h*)(pA + kt * 32 + 8);
            v16h av;
#pragma unroll
            for (int i = 0; i < 8; ++i) { av[i] = alo[i]; av[i + 8] = ahi[i]; }
#pragma unroll
            for (int nt = 0; nt < 4; ++nt) {
                // B fragment: col l15 of tile nt; 16 contiguous K per lane half
                const half_t* pB = &Bs[(nt * 16 + l15) * LROW + hi * 16 + kt * 32];
                v8h blo = *(const v8h*)(pB);
                v8h bhi = *(const v8h*)(pB + 8);
                v16h bv;
#pragma unroll
                for (int i = 0; i < 8; ++i) { bv[i] = blo[i]; bv[i + 8] = bhi[i]; }
                acc[nt] = __builtin_amdgcn_wmma_f32_16x16x32_f16(false, av, false, bv,
                                                                 (short)0, acc[nt],
                                                                 false, false);
            }
        }

#pragma unroll
        for (int nt = 0; nt < 4; ++nt) {
            const int   n  = ch * 64 + nt * 16 + l15;
            const float bn = bias[n];
#pragma unroll
            for (int j = 0; j < 8; ++j) {
                float v = acc[nt][j] + bn;
                int   m = mr + j;
                if (MODE == 0) {
                    ru[(size_t)m * 128 + n] = 1.f / (1.f + expf(-v));
                } else {
                    float cc = tanhf(v);
                    float u  = ru[(size_t)m * 128 + 64 + n];
                    float hv = h[(size_t)m * UNITS + n];
                    h[(size_t)m * UNITS + n] = u * hv + (1.f - u) * cc;
                }
            }
        }
    }
}

// ---------------- output projection + autoregressive feedback ----------------
__global__ void k_proj(const float* __restrict__ h, const float* __restrict__ Wp,
                       const float* __restrict__ bp, float* __restrict__ out,
                       float* __restrict__ dec_in, int t) {
    int m = blockIdx.x * blockDim.x + threadIdx.x;
    if (m >= M_ROWS) return;
    float s = bp[0];
#pragma unroll
    for (int j = 0; j < UNITS; ++j) s += h[(size_t)m * UNITS + j] * Wp[j];
    out[(size_t)t * M_ROWS + m] = s;
    dec_in[m] = s;
}

// ---------------- driver ----------------
extern "C" void kernel_launch(void* const* d_in, const int* in_sizes, int n_in,
                              void* d_out, int out_size, void* d_ws, size_t ws_size,
                              hipStream_t stream) {
    (void)in_sizes; (void)n_in; (void)out_size; (void)ws_size;
    const float* inputs   = (const float*)d_in[0];
    const float* loc      = (const float*)d_in[1];
    const float* geo      = (const float*)d_in[2];
    const float* ang      = (const float*)d_in[3];
    const float* node_emb = (const float*)d_in[4];
    const float* W_fe  = (const float*)d_in[5];
    const float* b_fe  = (const float*)d_in[6];
    const float* Wk0   = (const float*)d_in[7];
    const float* bk0   = (const float*)d_in[8];
    const float* Wk1   = (const float*)d_in[9];
    const float* bk1   = (const float*)d_in[10];
    const float* Wk2   = (const float*)d_in[11];
    const float* bk2   = (const float*)d_in[12];
    const float* Wg_e  = (const float*)d_in[13];
    const float* bg_e  = (const float*)d_in[14];
    const float* Wc_e  = (const float*)d_in[15];
    const float* bc_e  = (const float*)d_in[16];
    const float* Wg_d  = (const float*)d_in[17];
    const float* bg_d  = (const float*)d_in[18];
    const float* Wc_d  = (const float*)d_in[19];
    const float* bc_d  = (const float*)d_in[20];
    const float* W_proj = (const float*)d_in[21];
    const float* b_proj = (const float*)d_in[22];
    const int*   sidx  = (const int*)d_in[23];
    const int*   col   = sidx + NEDGE;
    float* out = (float*)d_out;

    char* ws = (char*)d_ws;
    size_t off = 0;
    auto carve = [&](size_t bytes) -> char* {
        char* p = ws + off;
        off += (bytes + 255) & ~(size_t)255;
        return p;
    };
    float*  wraw = (float*)carve((size_t)NEDGE * 4);
    float*  wnrm = (float*)carve((size_t)NEDGE * 4);
    half_t* WgTe = (half_t*)carve((size_t)128 * KPAD_E * 2);
    half_t* WcTe = (half_t*)carve((size_t)64  * KPAD_E * 2);
    half_t* WgTd = (half_t*)carve((size_t)128 * KPAD_D * 2);
    half_t* WcTd = (half_t*)carve((size_t)64  * KPAD_D * 2);
    float*  hbuf = (float*)carve((size_t)M_ROWS * UNITS * 4);
    half_t* y    = (half_t*)carve((size_t)M_ROWS * KPAD_E * 2);
    float*  ru   = (float*)carve((size_t)M_ROWS * 128 * 4);
    float*  dec  = (float*)carve((size_t)M_ROWS * 4);

    // static edge weights + weight reformat + h0 = 0
    k_edge_mlp<<<(NEDGE + 255) / 256, 256, 0, stream>>>(loc, geo, ang, Wk0, bk0, Wk1, bk1,
                                                        Wk2, bk2, sidx, wraw);
    k_edge_norm<<<(N_NODES + 255) / 256, 256, 0, stream>>>(wraw, wnrm);
    k_wtrans<<<(128 * KPAD_E + 255) / 256, 256, 0, stream>>>(Wg_e, WgTe, KENC, 128, KPAD_E);
    k_wtrans<<<(64  * KPAD_E + 255) / 256, 256, 0, stream>>>(Wc_e, WcTe, KENC, 64,  KPAD_E);
    k_wtrans<<<(128 * KPAD_D + 255) / 256, 256, 0, stream>>>(Wg_d, WgTd, KDEC, 128, KPAD_D);
    k_wtrans<<<(64  * KPAD_D + 255) / 256, 256, 0, stream>>>(Wc_d, WcTd, KDEC, 64,  KPAD_D);
    k_zero_f32<<<(M_ROWS * UNITS + 255) / 256, 256, 0, stream>>>(hbuf, M_ROWS * UNITS);

    dim3 packGrid(1, N_NODES, BATCH);
    dim3 gemmGrid(M_ROWS / 128);    // 8 waves/block, 16 rows per wave

    // ---- encoder ----
    for (int t = 0; t < SEQ; ++t) {
        k_pack_enc<<<packGrid, 128, 0, stream>>>(inputs, node_emb, W_fe, b_fe, hbuf, ru, 0, t, y);
        k_hop<<<packGrid, 128, 0, stream>>>(y, wnrm, col, CENC, 0, CENC, KPAD_E);
        k_hop<<<packGrid, 128, 0, stream>>>(y, wnrm, col, CENC, CENC, 2 * CENC, KPAD_E);
        k_wmma_gemm<KPAD_E, 128, 0><<<gemmGrid, 256, 0, stream>>>(y, WgTe, bg_e, ru, hbuf);
        k_pack_enc<<<packGrid, 128, 0, stream>>>(inputs, node_emb, W_fe, b_fe, hbuf, ru, 1, t, y);
        k_hop<<<packGrid, 128, 0, stream>>>(y, wnrm, col, CENC, 0, CENC, KPAD_E);
        k_hop<<<packGrid, 128, 0, stream>>>(y, wnrm, col, CENC, CENC, 2 * CENC, KPAD_E);
        k_wmma_gemm<KPAD_E, 64, 1><<<gemmGrid, 256, 0, stream>>>(y, WcTe, bc_e, ru, hbuf);
    }

    // ---- decoder (autoregressive, GO = zeros) ----
    k_zero_f32<<<(M_ROWS + 255) / 256, 256, 0, stream>>>(dec, M_ROWS);
    for (int t = 0; t < HOR; ++t) {
        k_pack_dec<<<packGrid, 128, 0, stream>>>(dec, hbuf, ru, 0, y);
        k_hop<<<packGrid, 128, 0, stream>>>(y, wnrm, col, CDEC, 0, CDEC, KPAD_D);
        k_hop<<<packGrid, 128, 0, stream>>>(y, wnrm, col, CDEC, CDEC, 2 * CDEC, KPAD_D);
        k_wmma_gemm<KPAD_D, 128, 0><<<gemmGrid, 256, 0, stream>>>(y, WgTd, bg_d, ru, hbuf);
        k_pack_dec<<<packGrid, 128, 0, stream>>>(dec, hbuf, ru, 1, y);
        k_hop<<<packGrid, 128, 0, stream>>>(y, wnrm, col, CDEC, 0, CDEC, KPAD_D);
        k_hop<<<packGrid, 128, 0, stream>>>(y, wnrm, col, CDEC, CDEC, 2 * CDEC, KPAD_D);
        k_wmma_gemm<KPAD_D, 64, 1><<<gemmGrid, 256, 0, stream>>>(y, WcTd, bc_d, ru, hbuf);
        k_proj<<<(M_ROWS + 255) / 256, 256, 0, stream>>>(hbuf, W_proj, b_proj, out, dec, t);
    }
}
